// SelfAttention_36584531427390
// MI455X (gfx1250) — matrix operand order repaired and checked by reading, not compiled, not run on hardware
//
#include <hip/hip_runtime.h>
#include <hip/hip_bf16.h>

// MI455X (gfx1250) self-attention, bf16 WMMA path with f32 softmax/accum.
// Phase 0: W -> bf16 transposed  Wt[3][HD][C]  (prep, L2-resident).
// Phase 1: QKV projection via v_wmma_f32_16x16x32_bf16, X staged in LDS as bf16.
// Phase 2: flash attention, K/V^T tiles in LDS, online softmax, WMMA for QK^T and PV.

typedef __attribute__((ext_vector_type(16))) __bf16 v16bf;
typedef __attribute__((ext_vector_type(8)))  float  v8f;
typedef __attribute__((ext_vector_type(2)))  __bf16 bf16x2;

#define B_N   8
#define T_SEQ 2048
#define C_DIM 1024
#define HD    128

// 16-bit A-matrix (16xK) VGPR packing: slot v in 0..7 packs K = kpairA(v,h), +1
__device__ __forceinline__ int kpairA(int v, int h) {
    return ((v >> 2) << 4) + (h << 3) + ((v & 3) << 1);
}
// 16-bit B-matrix (32x16) packing: same K order as A (slot v packs K = kpairA(v,h), +1)
__device__ __forceinline__ int kpairB(int v, int h) {
    return ((v >> 2) << 4) + (h << 3) + ((v & 3) << 1);
}

// ---------------- Phase 0: convert+transpose weights to bf16 ----------------
// Wt layout: [3][HD][C_DIM] bf16, so B-operand K-pairs (c, c+1) are contiguous.
__global__ __launch_bounds__(256) void w_prep_kernel(
    const float* __restrict__ Wq, const float* __restrict__ Wk,
    const float* __restrict__ Wv, __bf16* __restrict__ Wt)
{
    const int per = C_DIM * HD;
    int idx = blockIdx.x * 256 + threadIdx.x;
    if (idx >= 3 * per) return;
    const int w = idx / per, r = idx % per;
    const int c = r / HD, h = r % HD;
    const float* src = (w == 0) ? Wq : (w == 1) ? Wk : Wv;
    Wt[(size_t)w * per + (size_t)h * C_DIM + c] = (__bf16)src[r];
}

// ---------------- Phase 1: Q = XWq, K = XWk, V^T = (XWv)^T ----------------
__device__ __forceinline__ v8f proj_tile(const __bf16* __restrict__ xs,
                                         const __bf16* __restrict__ Wth, // [HD][C_DIM]
                                         int lm, int lh, int hbase)
{
    v8f acc = {};
    const __bf16* wrow = Wth + (size_t)(hbase + lm) * C_DIM;
    const __bf16* xrow = xs + lm * C_DIM;
    #pragma unroll 4
    for (int c = 0; c < C_DIM; c += 32) {
        v16bf a, bfr;
        #pragma unroll
        for (int v = 0; v < 8; ++v) {          // A: X rows (lane = row lm), from LDS
            const bf16x2 p = *(const bf16x2*)(xrow + c + kpairA(v, lh));
            a[2 * v] = p[0];  a[2 * v + 1] = p[1];
        }
        #pragma unroll
        for (int v = 0; v < 8; ++v) {          // B: Wt row, contiguous pairs
            const bf16x2 p = *(const bf16x2*)(wrow + c + kpairB(v, lh));
            bfr[2 * v] = p[0];  bfr[2 * v + 1] = p[1];
        }
        acc = __builtin_amdgcn_wmma_f32_16x16x32_bf16(false, a, false, bfr,
                                                      (short)0, acc, false, false);
    }
    return acc;
}

__global__ __launch_bounds__(256) void qkv_proj_kernel(
    const float* __restrict__ x, const __bf16* __restrict__ Wt,
    __bf16* __restrict__ Qo, __bf16* __restrict__ Ko, __bf16* __restrict__ VTo)
{
    __shared__ __bf16 xs[16 * C_DIM];          // 32 KB: 16-row X strip as bf16

    const int blk = blockIdx.x;                // b*128 + rowTile
    const int b   = blk >> 7;
    const int t0  = (blk & 127) << 4;
    const int tid = threadIdx.x;

    const float* xrow = x + ((size_t)b * T_SEQ + t0) * C_DIM;
    for (int i = tid * 4; i < 16 * C_DIM; i += 256 * 4) {
        const float4 f = *(const float4*)(xrow + i);
        xs[i + 0] = (__bf16)f.x;  xs[i + 1] = (__bf16)f.y;
        xs[i + 2] = (__bf16)f.z;  xs[i + 3] = (__bf16)f.w;
    }
    __syncthreads();

    const int lane  = tid & 31;
    const int wave  = tid >> 5;                // 8 waves -> 8 dim-tiles of 16
    const int hbase = wave << 4;
    const int lm    = lane & 15;
    const int lh    = lane >> 4;
    const int per   = C_DIM * HD;
    const int dim   = hbase + lm;

    // ---- Q ----
    {
        const v8f acc = proj_tile(xs, Wt + 0 * per, lm, lh, hbase);
        #pragma unroll
        for (int v = 0; v < 8; ++v) {
            const int row = t0 + v + 8 * lh;
            Qo[((size_t)b * T_SEQ + row) * HD + dim] = (__bf16)acc[v];
        }
    }
    // ---- K ----
    {
        const v8f acc = proj_tile(xs, Wt + 1 * per, lm, lh, hbase);
        #pragma unroll
        for (int v = 0; v < 8; ++v) {
            const int row = t0 + v + 8 * lh;
            Ko[((size_t)b * T_SEQ + row) * HD + dim] = (__bf16)acc[v];
        }
    }
    // ---- V (stored transposed [HD][T]) ----
    {
        const v8f acc = proj_tile(xs, Wt + 2 * per, lm, lh, hbase);
        #pragma unroll
        for (int v = 0; v < 8; ++v) {
            const int row = t0 + v + 8 * lh;
            VTo[((size_t)b * HD + dim) * T_SEQ + row] = (__bf16)acc[v];
        }
    }
}

// ---------------- Phase 2: causal flash attention ----------------
#define AW 4   // waves per block -> 64 query rows per block

__global__ __launch_bounds__(AW * 32) void attn_kernel(
    const __bf16* __restrict__ Q, const __bf16* __restrict__ K,
    const __bf16* __restrict__ VT, float* __restrict__ out)
{
    __shared__ __bf16 Kt[32 * HD];             // 8 KB  [key][dim]
    __shared__ __bf16 Vt[HD * 32];             // 8 KB  [dim][key]
    __shared__ __bf16 Pt[AW][16 * 32];         // 4 KB  per-wave P tile [row][key]

    const int blk  = blockIdx.x;               // b*(T/64) + qTile
    const int b    = blk >> 5;
    const int q0   = (blk & 31) << 6;          // 64-query block base
    const int tid  = threadIdx.x;
    const int lane = tid & 31;
    const int wave = tid >> 5;
    const int lm   = lane & 15;
    const int lh   = lane >> 4;
    const int qrow = q0 + wave * 16;           // wave's first query row
    const int nthr = AW * 32;

    // Q fragments (A layout), loaded once: 16 rows x 128 dims
    v16bf qf[4];
    const __bf16* Qbase = Q + ((size_t)b * T_SEQ + qrow + lm) * HD;
    #pragma unroll
    for (int kc = 0; kc < 4; ++kc)
        #pragma unroll
        for (int v = 0; v < 8; ++v) {
            const int kk = kc * 32 + kpairA(v, lh);
            const bf16x2 p = *(const bf16x2*)(Qbase + kk);
            qf[kc][2 * v] = p[0];  qf[kc][2 * v + 1] = p[1];
        }

    v8f   O[8] = {};
    float mrow[8], lrow[8];
    #pragma unroll
    for (int v = 0; v < 8; ++v) { mrow[v] = -__builtin_inff(); lrow[v] = 0.f; }

    const __bf16* Kb = K  + (size_t)b * T_SEQ * HD;
    const __bf16* Vb = VT + (size_t)b * HD * T_SEQ;
    const float   scl = 0.08838834764831845f;  // 1/sqrt(128)
    const int     nKB = (q0 >> 5) + 2;         // 32-key blocks covering [0, q0+64)

    for (int kb = 0; kb < nKB; ++kb) {
        __syncthreads();                       // protect tiles vs previous iteration
        {
            const __bf16* src = Kb + (size_t)(kb * 32) * HD;
            for (int i = tid * 2; i < 32 * HD; i += nthr * 2)
                *(bf16x2*)&Kt[i] = *(const bf16x2*)(src + i);
            for (int i = tid * 2; i < HD * 32; i += nthr * 2) {
                const int dim = i >> 5, key = i & 31;
                *(bf16x2*)&Vt[dim * 32 + key] =
                    *(const bf16x2*)(Vb + (size_t)dim * T_SEQ + kb * 32 + key);
            }
        }
        __syncthreads();

        // S = Q K^T : two 16-key sub-tiles, 4 dim-chunks each
        v8f S[2];
        #pragma unroll
        for (int nb = 0; nb < 2; ++nb) {
            v8f acc = {};
            const int key = nb * 16 + lm;
            #pragma unroll
            for (int kc = 0; kc < 4; ++kc) {
                v16bf bfr;
                #pragma unroll
                for (int v = 0; v < 8; ++v) {
                    const int kk = kc * 32 + kpairB(v, lh);
                    const bf16x2 p = *(const bf16x2*)(&Kt[key * HD + kk]);
                    bfr[2 * v] = p[0];  bfr[2 * v + 1] = p[1];
                }
                acc = __builtin_amdgcn_wmma_f32_16x16x32_bf16(false, qf[kc], false, bfr,
                                                              (short)0, acc, false, false);
            }
            S[nb] = acc;
        }

        // causal mask + online softmax (row stats via 16-lane xor reductions)
        float pscale[8];
        #pragma unroll
        for (int v = 0; v < 8; ++v) {
            const int row = qrow + v + 8 * lh;
            const int k0  = kb * 32 + lm;
            const int k1  = k0 + 16;
            float s0 = (k0 <= row) ? S[0][v] * scl : -__builtin_inff();
            float s1 = (k1 <= row) ? S[1][v] * scl : -__builtin_inff();
            float rm = fmaxf(s0, s1);
            #pragma unroll
            for (int m = 1; m < 16; m <<= 1) rm = fmaxf(rm, __shfl_xor(rm, m, 32));
            const float mnew = fmaxf(mrow[v], rm);
            const float sc   = __expf(mrow[v] - mnew);
            const float p0   = __expf(s0 - mnew);
            const float p1   = __expf(s1 - mnew);
            float rs = p0 + p1;
            #pragma unroll
            for (int m = 1; m < 16; m <<= 1) rs += __shfl_xor(rs, m, 32);
            lrow[v]   = lrow[v] * sc + rs;
            mrow[v]   = mnew;
            pscale[v] = sc;
            S[0][v] = p0;  S[1][v] = p1;
        }
        #pragma unroll
        for (int nd = 0; nd < 8; ++nd)
            #pragma unroll
            for (int v = 0; v < 8; ++v) O[nd][v] *= pscale[v];

        // P: C/D layout -> [row][key] tile in LDS (transpose for A-operand)
        #pragma unroll
        for (int nb = 0; nb < 2; ++nb)
            #pragma unroll
            for (int v = 0; v < 8; ++v)
                Pt[wave][(v + 8 * lh) * 32 + nb * 16 + lm] = (__bf16)S[nb][v];
        __syncthreads();

        // O += P * V : 8 dim-tiles of 16
        v16bf pf;
        #pragma unroll
        for (int v = 0; v < 8; ++v) {
            const int kk = kpairA(v, lh);
            const bf16x2 p = *(const bf16x2*)(&Pt[wave][lm * 32 + kk]);
            pf[2 * v] = p[0];  pf[2 * v + 1] = p[1];
        }
        #pragma unroll
        for (int nd = 0; nd < 8; ++nd) {
            v16bf bfr;
            const int dim = nd * 16 + lm;
            #pragma unroll
            for (int v = 0; v < 8; ++v) {
                const int kk = kpairB(v, lh);
                const bf16x2 p = *(const bf16x2*)(&Vt[dim * 32 + kk]);
                bfr[2 * v] = p[0];  bfr[2 * v + 1] = p[1];
            }
            O[nd] = __builtin_amdgcn_wmma_f32_16x16x32_bf16(false, pf, false, bfr,
                                                            (short)0, O[nd], false, false);
        }
    }

    // epilogue: out = O / l  (f32)
    #pragma unroll
    for (int nd = 0; nd < 8; ++nd) {
        const int dim = nd * 16 + lm;
        #pragma unroll
        for (int v = 0; v < 8; ++v) {
            const int row = qrow + v + 8 * lh;
            out[((size_t)b * T_SEQ + row) * HD + dim] = O[nd][v] / lrow[v];
        }
    }
}

extern "C" void kernel_launch(void* const* d_in, const int* in_sizes, int n_in,
                              void* d_out, int out_size, void* d_ws, size_t ws_size,
                              hipStream_t stream) {
    const float* x  = (const float*)d_in[0];
    const float* Wq = (const float*)d_in[1];
    const float* Wk = (const float*)d_in[2];
    const float* Wv = (const float*)d_in[3];
    float* out = (float*)d_out;

    // Workspace: Q, K (row-major), V^T (bf16, 12 MB) + Wt bf16 transposed (768 KB).
    const size_t elems = (size_t)B_N * T_SEQ * HD;
    __bf16* Qw  = (__bf16*)d_ws;
    __bf16* Kw  = Qw + elems;
    __bf16* VTw = Kw + elems;
    __bf16* Wt  = VTw + elems;

    const int wtotal = 3 * C_DIM * HD;
    w_prep_kernel<<<(wtotal + 255) / 256, 256, 0, stream>>>(Wq, Wk, Wv, Wt);
    qkv_proj_kernel<<<B_N * (T_SEQ / 16), 256, 0, stream>>>(x, Wt, Qw, Kw, VTw);
    attn_kernel<<<B_N * (T_SEQ / 64), AW * 32, 0, stream>>>(Qw, Kw, VTw, out);
}
